// MAE3D_8933531976108
// MI455X (gfx1250) — compile-verified, matrix operand order checked by reading, not run on hardware
//
#include <hip/hip_runtime.h>
#include <hip/hip_bf16.h>

#define BN 4
#define GG 60
#define PP 5
#define NPDV 12
#define NPV 1728
#define EV 256
#define HV 8
#define SVV 432
#define NMASK 1296
#define NCV 40
#define P3V 125

typedef _Float16 h8t  __attribute__((ext_vector_type(8)));
typedef _Float16 h16t __attribute__((ext_vector_type(16)));
typedef float    f8t  __attribute__((ext_vector_type(8)));

// ---------------------------------------------------------------------------
// WMMA GEMM, 16(M) x 64(N) per wave: one A fragment is reused across four
// B fragments -> 4 v_wmma per A load (1.6x fewer loads per WMMA than a
// 16x16 tile). OOB rows/cols are CLAMPED (not branched) so the inner loop
// has no exec-mask toggles; garbage tiles are masked at the store.
// C[M,N] = act(A[M,Kp] @ W[N,Kp]^T + bias + pos) + residual, with optional
// row gather (rowidx) and f32/f16 outputs.
// ---------------------------------------------------------------------------
__global__ void gemm_wmma_f16(const _Float16* __restrict__ A, const _Float16* __restrict__ W,
                              const float* __restrict__ bias, const float* __restrict__ pos,
                              const int* __restrict__ rowidx, const int* __restrict__ posidx,
                              const float* __restrict__ residual,
                              float* __restrict__ C, _Float16* __restrict__ C16,
                              int M, int N, int Kp, int ldc, int act)
{
    const int wave = (int)((blockIdx.x * blockDim.x + threadIdx.x) >> 5);
    const int lane = threadIdx.x & 31;
    const int tilesN = (N + 63) >> 6;
    const int tilesM = (M + 15) >> 4;
    if (wave >= tilesM * tilesN) return;           // wave-uniform: EXEC stays full
    const int tm = wave / tilesN;
    const int tn = wave - tm * tilesN;
    const int lm = lane & 15;
    const int hi = lane >> 4;

    // A row (clamped, optional gather)
    const int am  = tm * 16 + lm;
    const int amc = (am < M) ? am : (M - 1);
    const int arow = rowidx ? rowidx[amc] : amc;
    const _Float16* aPtr = A + (long)arow * Kp + hi * 8;

    // Four B column groups (clamped)
    const int n0 = tn * 64;
    const _Float16* bPtr0;
    const _Float16* bPtr1;
    const _Float16* bPtr2;
    const _Float16* bPtr3;
    {
        int b0 = n0 + lm;        if (b0 > N - 1) b0 = N - 1;
        int b1 = n0 + 16 + lm;   if (b1 > N - 1) b1 = N - 1;
        int b2 = n0 + 32 + lm;   if (b2 > N - 1) b2 = N - 1;
        int b3 = n0 + 48 + lm;   if (b3 > N - 1) b3 = N - 1;
        bPtr0 = W + (long)b0 * Kp + hi * 8;
        bPtr1 = W + (long)b1 * Kp + hi * 8;
        bPtr2 = W + (long)b2 * Kp + hi * 8;
        bPtr3 = W + (long)b3 * Kp + hi * 8;
    }

    f8t acc0, acc1, acc2, acc3;
#pragma unroll
    for (int i = 0; i < 8; ++i) { acc0[i] = 0.f; acc1[i] = 0.f; acc2[i] = 0.f; acc3[i] = 0.f; }

    for (int k0 = 0; k0 < Kp; k0 += 32) {
        // stream prefetch one K-step ahead (global_prefetch_b8)
        __builtin_prefetch(aPtr + k0 + 32, 0, 3);
        __builtin_prefetch(bPtr0 + k0 + 32, 0, 3);

        h8t alo = *(const h8t*)(aPtr + k0);
        h8t ahi = *(const h8t*)(aPtr + k0 + 16);
        h16t a = __builtin_shufflevector(alo, ahi, 0,1,2,3,4,5,6,7,8,9,10,11,12,13,14,15);

        h8t l0 = *(const h8t*)(bPtr0 + k0);
        h8t h0 = *(const h8t*)(bPtr0 + k0 + 16);
        h8t l1 = *(const h8t*)(bPtr1 + k0);
        h8t h1 = *(const h8t*)(bPtr1 + k0 + 16);
        h8t l2 = *(const h8t*)(bPtr2 + k0);
        h8t h2 = *(const h8t*)(bPtr2 + k0 + 16);
        h8t l3 = *(const h8t*)(bPtr3 + k0);
        h8t h3 = *(const h8t*)(bPtr3 + k0 + 16);

        h16t b0 = __builtin_shufflevector(l0, h0, 0,1,2,3,4,5,6,7,8,9,10,11,12,13,14,15);
        h16t b1 = __builtin_shufflevector(l1, h1, 0,1,2,3,4,5,6,7,8,9,10,11,12,13,14,15);
        h16t b2 = __builtin_shufflevector(l2, h2, 0,1,2,3,4,5,6,7,8,9,10,11,12,13,14,15);
        h16t b3 = __builtin_shufflevector(l3, h3, 0,1,2,3,4,5,6,7,8,9,10,11,12,13,14,15);

        acc0 = __builtin_amdgcn_wmma_f32_16x16x32_f16(false, a, false, b0, (short)0, acc0, false, false);
        acc1 = __builtin_amdgcn_wmma_f32_16x16x32_f16(false, a, false, b1, (short)0, acc1, false, false);
        acc2 = __builtin_amdgcn_wmma_f32_16x16x32_f16(false, a, false, b2, (short)0, acc2, false, false);
        acc3 = __builtin_amdgcn_wmma_f32_16x16x32_f16(false, a, false, b3, (short)0, acc3, false, false);
    }

    // epilogue: 4 subtiles, bounds-checked stores
    f8t* accs[4] = { &acc0, &acc1, &acc2, &acc3 };
#pragma unroll
    for (int j = 0; j < 4; ++j) {
        const int n = n0 + j * 16 + lm;
        if (n < N) {
            const float bs = bias ? bias[n] : 0.f;
#pragma unroll
            for (int v = 0; v < 8; ++v) {
                const int m = tm * 16 + v + hi * 8;
                if (m < M) {
                    float val = (*accs[j])[v] + bs;
                    if (pos)      val += pos[(long)posidx[m] * N + n];
                    if (act)      val = 0.5f * val * (1.f + erff(val * 0.70710678118654752f));
                    if (residual) val += residual[(long)m * ldc + n];
                    if (C)   C[(long)m * ldc + n]   = val;
                    if (C16) C16[(long)m * ldc + n] = (_Float16)val;
                }
            }
        }
    }
}

// ---------------------------------------------------------------------------
// Flash attention: one wave per (b, h, 16-row q tile). dh=32 -> one WMMA per
// 16-key score tile; P transposed D-layout -> A-layout through per-wave LDS
// with explicit s_wait_dscnt; P@V via K=32 WMMA with upper 16 k-halves zeroed.
// qkv layout: (B*S, 768) f16 with q|k|v at col offsets 0|256|512, head h at +h*32.
// ---------------------------------------------------------------------------
__global__ void attn_flash(const _Float16* __restrict__ qkv, _Float16* __restrict__ out, int S)
{
    __shared__ _Float16 plds[8][16 * 32];
    const int wid  = threadIdx.x >> 5;
    const int lane = threadIdx.x & 31;
    const int qtiles = S >> 4;
    const int wave = blockIdx.x * 8 + wid;
    if (wave >= BN * HV * qtiles) return;          // wave-uniform
    const int qt = wave % qtiles;
    const int h  = (wave / qtiles) % HV;
    const int b  = wave / (qtiles * HV);
    const int lm = lane & 15;
    const int hi = lane >> 4;

    const _Float16* Qp = qkv + ((long)(b * S + qt * 16 + lm)) * 768 + h * 32;
    h8t qlo = *(const h8t*)(Qp + hi * 8);
    h8t qhi = *(const h8t*)(Qp + 16 + hi * 8);
    h16t qa = __builtin_shufflevector(qlo, qhi, 0,1,2,3,4,5,6,7,8,9,10,11,12,13,14,15);

    f8t o0, o1;
    float rmax[8], rsum[8];
#pragma unroll
    for (int v = 0; v < 8; ++v) { o0[v] = 0.f; o1[v] = 0.f; rmax[v] = -1e30f; rsum[v] = 0.f; }

    // zero k = 16..31 columns of the P slab once (stay zero every iteration)
#pragma unroll
    for (int v = 0; v < 8; ++v) plds[wid][(v + hi * 8) * 32 + 16 + lm] = (_Float16)0.f;

    const float sc = 0.17677669529663688f;   // 1/sqrt(32)

    for (int kc = 0; kc < S; kc += 16) {
        // scores: A = Q (16x32), B = K-chunk^T (32x16), lane col = key lm
        const _Float16* Kp = qkv + ((long)(b * S + kc + lm)) * 768 + 256 + h * 32;
        h8t klo = *(const h8t*)(Kp + hi * 8);
        h8t khi = *(const h8t*)(Kp + 16 + hi * 8);
        h16t kb = __builtin_shufflevector(klo, khi, 0,1,2,3,4,5,6,7,8,9,10,11,12,13,14,15);
        f8t zc;
#pragma unroll
        for (int i = 0; i < 8; ++i) zc[i] = 0.f;
        f8t s = __builtin_amdgcn_wmma_f32_16x16x32_f16(false, qa, false, kb, (short)0, zc, false, false);

        // online softmax per row (row m = v + hi*8 spread across 16 lanes of the half)
#pragma unroll
        for (int v = 0; v < 8; ++v) {
            float x = s[v] * sc;
            float mx = x;
            mx = fmaxf(mx, __shfl_xor(mx, 1, 32));
            mx = fmaxf(mx, __shfl_xor(mx, 2, 32));
            mx = fmaxf(mx, __shfl_xor(mx, 4, 32));
            mx = fmaxf(mx, __shfl_xor(mx, 8, 32));
            float nmax  = fmaxf(rmax[v], mx);
            float scale = __expf(rmax[v] - nmax);
            float p     = __expf(x - nmax);
            float ps = p;
            ps += __shfl_xor(ps, 1, 32);
            ps += __shfl_xor(ps, 2, 32);
            ps += __shfl_xor(ps, 4, 32);
            ps += __shfl_xor(ps, 8, 32);
            rsum[v] = rsum[v] * scale + ps;
            rmax[v] = nmax;
            o0[v] *= scale;
            o1[v] *= scale;
            plds[wid][(v + hi * 8) * 32 + lm] = (_Float16)p;   // (m, k) slab
        }

        // D-layout -> A-layout transpose through LDS (cross-lane; force the wait)
        asm volatile("s_wait_dscnt 0" ::: "memory");
        const _Float16* pp = &plds[wid][lm * 32];
        h8t plo = *(const h8t*)(pp + hi * 8);
        h8t phi = *(const h8t*)(pp + 16 + hi * 8);
        h16t pa = __builtin_shufflevector(plo, phi, 0,1,2,3,4,5,6,7,8,9,10,11,12,13,14,15);

        // V chunk as B (32x16 per dh-subtile); k rows 16..31 hard-zeroed
        h16t vb0, vb1;
#pragma unroll
        for (int j = 0; j < 8; ++j) {
            const _Float16* Vp = qkv + ((long)(b * S + kc + hi * 8 + j)) * 768 + 512 + h * 32;
            vb0[j]     = Vp[lm];
            vb1[j]     = Vp[16 + lm];
            vb0[8 + j] = (_Float16)0.f;
            vb1[8 + j] = (_Float16)0.f;
        }
        o0 = __builtin_amdgcn_wmma_f32_16x16x32_f16(false, pa, false, vb0, (short)0, o0, false, false);
        o1 = __builtin_amdgcn_wmma_f32_16x16x32_f16(false, pa, false, vb1, (short)0, o1, false, false);
    }

#pragma unroll
    for (int v = 0; v < 8; ++v) {
        const int m = qt * 16 + v + hi * 8;
        const float inv = 1.f / rsum[v];
        _Float16* op = out + ((long)(b * S + m)) * 256 + h * 32;
        op[lm]      = (_Float16)(o0[v] * inv);
        op[16 + lm] = (_Float16)(o1[v] * inv);
    }
}

// ---------------------------------------------------------------------------
// LayerNorm over E=256, one wave per row, wave32 shfl_xor reductions.
// ---------------------------------------------------------------------------
__global__ void ln_rows(const float* __restrict__ X, const float* __restrict__ w,
                        const float* __restrict__ bb, float* __restrict__ Y,
                        _Float16* __restrict__ Y16, int M)
{
    const int wave = (int)((blockIdx.x * blockDim.x + threadIdx.x) >> 5);
    const int lane = threadIdx.x & 31;
    if (wave >= M) return;
    const float* x = X + (long)wave * EV;
    float v[8];
    float s = 0.f;
#pragma unroll
    for (int i = 0; i < 8; ++i) { v[i] = x[lane + 32 * i]; s += v[i]; }
    for (int d = 1; d < 32; d <<= 1) s += __shfl_xor(s, d, 32);
    const float mean = s * (1.f / 256.f);
    float var = 0.f;
#pragma unroll
    for (int i = 0; i < 8; ++i) { float dv = v[i] - mean; var += dv * dv; }
    for (int d = 1; d < 32; d <<= 1) var += __shfl_xor(var, d, 32);
    var *= (1.f / 256.f);
    const float inv = rsqrtf(var + 1e-5f);
#pragma unroll
    for (int i = 0; i < 8; ++i) {
        const int c = lane + 32 * i;
        const float y = (v[i] - mean) * inv * w[c] + bb[c];
        Y[(long)wave * EV + c] = y;
        if (Y16) Y16[(long)wave * EV + c] = (_Float16)y;
    }
}

// ---------------------------------------------------------------------------
// Patchify (B,1,60,60,60) -> xg f32 (B,NP,125) + f16 copy padded to K=128.
// ---------------------------------------------------------------------------
__global__ void patchify(const float* __restrict__ x, float* __restrict__ xg,
                         _Float16* __restrict__ xg16)
{
    long idx = (long)blockIdx.x * blockDim.x + threadIdx.x;
    if (idx >= (long)BN * NPV * P3V) return;
    int e = (int)(idx % P3V);
    long rp = idx / P3V;
    int p = (int)(rp % NPV);
    int b = (int)(rp / NPV);
    int pc = p % NPDV; int t = p / NPDV; int pb = t % NPDV; int pa = t / NPDV;
    int ec = e % PP;  int t2 = e / PP;  int eb = t2 % PP;  int ea = t2 / PP;
    int gz = pa * PP + ea, gy = pb * PP + eb, gx = pc * PP + ec;
    float val = x[(((long)b * GG + gz) * GG + gy) * GG + gx];
    xg[idx] = val;
    xg16[rp * 128 + e] = (_Float16)val;
    if (e < 3) xg16[rp * 128 + 125 + e] = (_Float16)0.f;
}

// ---------------------------------------------------------------------------
// Stable-argsort ranking per batch (matches jnp.argsort tie-break), builds
// mask, keep lists (global row + patch id) and inverse scatter map.
// ---------------------------------------------------------------------------
__global__ void rank_mask(const float* __restrict__ noise, int* __restrict__ keeprow,
                          int* __restrict__ keepid, int* __restrict__ inv,
                          float* __restrict__ mask)
{
    const int b = blockIdx.x;
    __shared__ float sn[NPV];
    for (int i = threadIdx.x; i < NPV; i += blockDim.x) sn[i] = noise[b * NPV + i];
    __syncthreads();
    for (int i = threadIdx.x; i < NPV; i += blockDim.x) {
        const float vi = sn[i];
        int r = 0;
        for (int j = 0; j < NPV; ++j) {
            const float vj = sn[j];
            r += (vj < vi) || (vj == vi && j < i);
        }
        mask[b * NPV + i] = (r < NMASK) ? 1.f : 0.f;
        if (r >= NMASK) {
            const int s = r - NMASK;
            keeprow[b * SVV + s] = b * NPV + i;
            keepid[b * SVV + s]  = i;
            inv[b * NPV + i] = s;
        } else {
            inv[b * NPV + i] = -1;
        }
    }
}

// ---------------------------------------------------------------------------
// Decoder input: mask_token / scattered-encoder + dec_pos_embed.
// ---------------------------------------------------------------------------
__global__ void dec_scatter(const float* __restrict__ enc, const int* __restrict__ inv,
                            const float* __restrict__ mtok, const float* __restrict__ dpos,
                            float* __restrict__ xd, _Float16* __restrict__ xd16)
{
    long idx = (long)blockIdx.x * blockDim.x + threadIdx.x;
    if (idx >= (long)BN * NPV * EV) return;
    int c = (int)(idx & 255);
    long row = idx >> 8;
    int p = (int)(row % NPV);
    int b = (int)(row / NPV);
    int s = inv[row];
    float v = (s >= 0) ? enc[((long)b * SVV + s) * EV + c] : mtok[c];
    v += dpos[(long)p * EV + c];
    xd[idx] = v;
    xd16[idx] = (_Float16)v;
}

// ---------------------------------------------------------------------------
// cls head: mean over S=432 then tiny (40x256) matvec; one block per batch.
// ---------------------------------------------------------------------------
__global__ void cls_head(const float* __restrict__ enc, const float* __restrict__ cw,
                         const float* __restrict__ cb, float* __restrict__ out)
{
    const int b = blockIdx.x;
    __shared__ float mean[EV];
    float acc = 0.f;
    for (int s = 0; s < SVV; ++s) acc += enc[((long)b * SVV + s) * EV + threadIdx.x];
    mean[threadIdx.x] = acc * (1.f / SVV);
    __syncthreads();
    if (threadIdx.x < NCV) {
        float o = cb[threadIdx.x];
        for (int c = 0; c < EV; ++c) o += mean[c] * cw[threadIdx.x * EV + c];
        out[b * NCV + threadIdx.x] = o;
    }
}

__global__ void cvt_f16(const float* __restrict__ src, _Float16* __restrict__ dst, long n)
{
    long i = (long)blockIdx.x * blockDim.x + threadIdx.x;
    if (i < n) dst[i] = (_Float16)src[i];
}

__global__ void cvt_pad_patch(const float* __restrict__ src, _Float16* __restrict__ dst)
{
    int i = blockIdx.x * blockDim.x + threadIdx.x;  // 256 rows * 128 padded cols
    if (i >= 256 * 128) return;
    int c = i & 127, r = i >> 7;
    dst[i] = (c < 125) ? (_Float16)src[r * 125 + c] : (_Float16)0.f;
}

extern "C" void kernel_launch(void* const* d_in, const int* in_sizes, int n_in,
                              void* d_out, int out_size, void* d_ws, size_t ws_size,
                              hipStream_t stream)
{
    (void)in_sizes; (void)n_in; (void)out_size; (void)ws_size;
    const float* x      = (const float*)d_in[0];
    const float* noise  = (const float*)d_in[1];

    float* out      = (float*)d_out;
    float* out_pred = out;                       // B*NP*125 = 864000
    float* out_xg   = out + 864000;              // 864000
    float* out_mask = out + 1728000;             // 6912
    float* out_cls  = out + 1734912;             // 160

    char* ws = (char*)d_ws;
    size_t off = 0;
    auto alloc = [&](size_t bytes) -> void* {
        void* p = ws + off;
        off += (bytes + 255) & ~(size_t)255;
        return p;
    };

    _Float16* w_patch = (_Float16*)alloc(256L * 128 * 2);
    _Float16* w_eqkv  = (_Float16*)alloc(4L * 768 * 256 * 2);
    _Float16* w_eout  = (_Float16*)alloc(4L * 256 * 256 * 2);
    _Float16* w_ef1   = (_Float16*)alloc(4L * 1024 * 256 * 2);
    _Float16* w_ef2   = (_Float16*)alloc(4L * 256 * 1024 * 2);
    _Float16* w_dqkv  = (_Float16*)alloc(2L * 768 * 256 * 2);
    _Float16* w_dout  = (_Float16*)alloc(2L * 256 * 256 * 2);
    _Float16* w_df1   = (_Float16*)alloc(2L * 1024 * 256 * 2);
    _Float16* w_df2   = (_Float16*)alloc(2L * 256 * 1024 * 2);
    _Float16* w_pred  = (_Float16*)alloc(125L * 256 * 2);

    _Float16* xg16   = (_Float16*)alloc((long)BN * NPV * 128 * 2);
    int* keeprow     = (int*)alloc(BN * SVV * 4);
    int* keepid      = (int*)alloc(BN * SVV * 4);
    int* inv         = (int*)alloc(BN * NPV * 4);

    float*    xf[3];
    _Float16* xh[3];
    for (int t = 0; t < 3; ++t) {
        xf[t] = (float*)alloc(6912L * 256 * 4);
        xh[t] = (_Float16*)alloc(6912L * 256 * 2);
    }
    float*    tmp    = (float*)alloc(6912L * 256 * 4);
    _Float16* qkv16  = (_Float16*)alloc(6912L * 768 * 2);
    _Float16* attn16 = (_Float16*)alloc(6912L * 256 * 2);
    _Float16* ffn16  = (_Float16*)alloc(6912L * 1024 * 2);

    auto CVT = [&](const void* s, _Float16* d, long n) {
        cvt_f16<<<(int)((n + 255) / 256), 256, 0, stream>>>((const float*)s, d, n);
    };
    cvt_pad_patch<<<128, 256, 0, stream>>>((const float*)d_in[2], w_patch);
    CVT(d_in[11], w_eqkv, 4L * 768 * 256);
    CVT(d_in[13], w_eout, 4L * 256 * 256);
    CVT(d_in[19], w_ef1,  4L * 1024 * 256);
    CVT(d_in[21], w_ef2,  4L * 256 * 1024);
    CVT(d_in[23], w_dqkv, 2L * 768 * 256);
    CVT(d_in[25], w_dout, 2L * 256 * 256);
    CVT(d_in[31], w_df1,  2L * 1024 * 256);
    CVT(d_in[33], w_df2,  2L * 256 * 1024);
    CVT(d_in[7],  w_pred, 125L * 256);

    patchify<<<(int)(((long)BN * NPV * P3V + 255) / 256), 256, 0, stream>>>(x, out_xg, xg16);
    rank_mask<<<BN, 256, 0, stream>>>(noise, keeprow, keepid, inv, out_mask);

    auto GEMM = [&](const _Float16* A, const _Float16* W, const float* bias, const float* pos,
                    const int* ridx, const int* pidx, const float* res,
                    float* C, _Float16* C16, int M, int N, int Kp, int ldc, int act) {
        long tiles = (long)((M + 15) / 16) * ((N + 63) / 64);
        int blocks = (int)((tiles + 7) / 8);
        gemm_wmma_f16<<<blocks, 256, 0, stream>>>(A, W, bias, pos, ridx, pidx, res,
                                                  C, C16, M, N, Kp, ldc, act);
    };

    // Token embed on visible patches only (gathered via rowidx), + patch_b + pos_embed
    GEMM(xg16, w_patch, (const float*)d_in[3], (const float*)d_in[4], keeprow, keepid,
         nullptr, xf[0], xh[0], BN * SVV, 256, 128, 256, 0);

    // -------- encoder (4 layers, S=432) --------
    const int Me = BN * SVV;
    for (int i = 0; i < 4; ++i) {
        GEMM(xh[0], w_eqkv + (long)i * 768 * 256, (const float*)d_in[12] + i * 768,
             nullptr, nullptr, nullptr, nullptr, nullptr, qkv16, Me, 768, 256, 768, 0);
        {
            int waves = BN * HV * (SVV / 16);
            attn_flash<<<(waves + 7) / 8, 256, 0, stream>>>(qkv16, attn16, SVV);
        }
        GEMM(attn16, w_eout + (long)i * 256 * 256, (const float*)d_in[14] + i * 256,
             nullptr, nullptr, nullptr, xf[0], tmp, nullptr, Me, 256, 256, 256, 0);
        ln_rows<<<(Me + 7) / 8, 256, 0, stream>>>(tmp, (const float*)d_in[15] + i * 256,
                                                  (const float*)d_in[16] + i * 256, xf[1], xh[1], Me);
        GEMM(xh[1], w_ef1 + (long)i * 1024 * 256, (const float*)d_in[20] + i * 1024,
             nullptr, nullptr, nullptr, nullptr, nullptr, ffn16, Me, 1024, 256, 1024, 1);
        GEMM(ffn16, w_ef2 + (long)i * 256 * 1024, (const float*)d_in[22] + i * 256,
             nullptr, nullptr, nullptr, xf[1], tmp, nullptr, Me, 256, 1024, 256, 0);
        ln_rows<<<(Me + 7) / 8, 256, 0, stream>>>(tmp, (const float*)d_in[17] + i * 256,
                                                  (const float*)d_in[18] + i * 256, xf[0], xh[0], Me);
    }

    cls_head<<<BN, 256, 0, stream>>>(xf[0], (const float*)d_in[9], (const float*)d_in[10], out_cls);

    // -------- decoder (2 layers, S=1728) --------
    dec_scatter<<<(int)(((long)BN * NPV * EV + 255) / 256), 256, 0, stream>>>(
        xf[0], inv, (const float*)d_in[6], (const float*)d_in[5], xf[2], xh[2]);
    const int Md = BN * NPV;
    for (int i = 0; i < 2; ++i) {
        GEMM(xh[2], w_dqkv + (long)i * 768 * 256, (const float*)d_in[24] + i * 768,
             nullptr, nullptr, nullptr, nullptr, nullptr, qkv16, Md, 768, 256, 768, 0);
        {
            int waves = BN * HV * (NPV / 16);
            attn_flash<<<(waves + 7) / 8, 256, 0, stream>>>(qkv16, attn16, NPV);
        }
        GEMM(attn16, w_dout + (long)i * 256 * 256, (const float*)d_in[26] + i * 256,
             nullptr, nullptr, nullptr, xf[2], tmp, nullptr, Md, 256, 256, 256, 0);
        ln_rows<<<(Md + 7) / 8, 256, 0, stream>>>(tmp, (const float*)d_in[27] + i * 256,
                                                  (const float*)d_in[28] + i * 256, xf[1], xh[1], Md);
        GEMM(xh[1], w_df1 + (long)i * 1024 * 256, (const float*)d_in[32] + i * 1024,
             nullptr, nullptr, nullptr, nullptr, nullptr, ffn16, Md, 1024, 256, 1024, 1);
        GEMM(ffn16, w_df2 + (long)i * 256 * 1024, (const float*)d_in[34] + i * 256,
             nullptr, nullptr, nullptr, xf[1], tmp, nullptr, Md, 256, 1024, 256, 0);
        ln_rows<<<(Md + 7) / 8, 256, 0, stream>>>(tmp, (const float*)d_in[29] + i * 256,
                                                  (const float*)d_in[30] + i * 256, xf[2], xh[2], Md);
    }

    // Pixel prediction head: N=125 (lane-guarded), writes straight into d_out
    GEMM(xh[2], w_pred, (const float*)d_in[8], nullptr, nullptr, nullptr, nullptr,
         out_pred, nullptr, Md, 125, 256, 125, 0);
}
